// CrossAttention_48842368090446
// MI455X (gfx1250) — matrix operand order repaired and checked by reading, not compiled, not run on hardware
//
#include <hip/hip_runtime.h>
#include <stdint.h>

// Problem dims (compile-time constants from the reference)
#define TB   2        // batch
#define TNQ  1024     // n  (queries)
#define TNK  4096     // N  (keys)
#define TC   768      // C
#define THH  12       // heads
#define TD   64       // head dim
#define TBH  (TB*THH) // 24

typedef __attribute__((ext_vector_type(16))) __bf16   v16bf;
typedef __attribute__((ext_vector_type(8)))  float    v8f;
typedef __attribute__((ext_vector_type(4)))  unsigned u32x4;
typedef __attribute__((ext_vector_type(2)))  unsigned u32x2;
typedef __attribute__((ext_vector_type(4)))  float    f32x4;

union Frag {
  v16bf v;
  u32x4 q[2];
  unsigned int u[8];
};

__device__ __forceinline__ unsigned short f32_to_bf16_bits(float f) {
  unsigned int x = __float_as_uint(f);
  x += 0x7FFFu + ((x >> 16) & 1u);          // round-to-nearest-even
  return (unsigned short)(x >> 16);
}
__device__ __forceinline__ unsigned int pack2_bf16(float lo, float hi) {
  return (unsigned int)f32_to_bf16_bits(lo) |
         ((unsigned int)f32_to_bf16_bits(hi) << 16);
}

// ---- WMMA operand loaders (wave32, 16x16x32 bf16 layouts per CDNA5 ISA) ----
// A (16xK, bf16): lane m<16 holds row m, K = k0+0..7 (VGPR0-3) and k0+16..23
// (VGPR4-7); lane m+16 holds row m, K = k0+8..15 and k0+24..31.
__device__ __forceinline__ v16bf load_A_bf16(const unsigned short* base, int ld,
                                             int row0, int k0, int lane) {
  const int r  = row0 + (lane & 15);
  const int kh = (lane & 16) ? 8 : 0;
  const unsigned short* p = base + (size_t)r * ld + k0 + kh;
  Frag f;
  f.q[0] = *(const u32x4*)(p);
  f.q[1] = *(const u32x4*)(p + 16);
  return f.v;
}

// Same A fragment, but source is f32 (attention probabilities) converted
// in-register to bf16.
__device__ __forceinline__ v16bf load_A_f32(const float* base, int ld,
                                            int row0, int k0, int lane) {
  const int r  = row0 + (lane & 15);
  const int kh = (lane & 16) ? 8 : 0;
  const float* p = base + (size_t)r * ld + k0 + kh;
  f32x4 a0 = *(const f32x4*)(p);
  f32x4 a1 = *(const f32x4*)(p + 4);
  f32x4 a2 = *(const f32x4*)(p + 16);
  f32x4 a3 = *(const f32x4*)(p + 20);
  Frag f;
  f.u[0] = pack2_bf16(a0.x, a0.y);
  f.u[1] = pack2_bf16(a0.z, a0.w);
  f.u[2] = pack2_bf16(a1.x, a1.y);
  f.u[3] = pack2_bf16(a1.z, a1.w);
  f.u[4] = pack2_bf16(a2.x, a2.y);
  f.u[5] = pack2_bf16(a2.z, a2.w);
  f.u[6] = pack2_bf16(a3.x, a3.y);
  f.u[7] = pack2_bf16(a3.z, a3.w);
  return f.v;
}

// B (Kx16, bf16), sourced from B^T stored row-major [Ncols, K]: lane n<16
// holds column n with K = k0+0..7 (VGPR0-3) and k0+16..23 (VGPR4-7); lane
// n+16 holds column n with K = k0+8..15 and k0+24..31.
__device__ __forceinline__ v16bf load_B_bf16(const unsigned short* baseT, int ld,
                                             int col0, int k0, int lane) {
  const int c  = col0 + (lane & 15);
  const int kh = (lane & 16) ? 8 : 0;
  const unsigned short* p = baseT + (size_t)c * ld + k0 + kh;
  Frag f;
  f.q[0] = *(const u32x4*)(p);
  f.q[1] = *(const u32x4*)(p + 16);
  return f.v;
}

__device__ __forceinline__ v8f wmma_bf16(v16bf a, v16bf b, v8f c) {
  return __builtin_amdgcn_wmma_f32_16x16x32_bf16(false, a, false, b, (short)0,
                                                 c, false, false);
}

// One wave computes a 16(M) x 64(N) tile: 1 A fragment, 4 B fragments,
// 4 f32 accumulators per 32-wide K step. Simple load-then-consume body:
// with compile-time K the compiler fully unrolls, clause-groups the b128
// loads, and pipelines with partial s_wait_loadcnt on its own.
template <bool A_IS_F32>
__device__ __forceinline__ void gemm_wave(const void* Abase, int lda,
                                          const unsigned short* Bt, int ldb,
                                          int m0, int n0, int K, int lane,
                                          v8f acc[4]) {
  for (int k0 = 0; k0 < K; k0 += 32) {
    v16bf a;
    if (A_IS_F32) a = load_A_f32((const float*)Abase, lda, m0, k0, lane);
    else          a = load_A_bf16((const unsigned short*)Abase, lda, m0, k0, lane);
    if (k0 + 32 < K) {   // hint next B chunk -> global_prefetch_b8
      __builtin_prefetch(Bt + (size_t)(n0 + (lane & 15)) * ldb + k0 + 32, 0, 1);
    }
#pragma unroll
    for (int t = 0; t < 4; ++t) {
      v16bf b = load_B_bf16(Bt, ldb, n0 + 16 * t, k0, lane);
      acc[t] = wmma_bf16(a, b, acc[t]);
    }
  }
}

#define ACC_INIT                                        \
  v8f zero = {0.f, 0.f, 0.f, 0.f, 0.f, 0.f, 0.f, 0.f};  \
  v8f acc[4] = {zero, zero, zero, zero};

// -------------------- conversion kernels --------------------
__global__ void __launch_bounds__(256)
k_cvt_bf16(const float* __restrict__ src, unsigned short* __restrict__ dst, int nelem) {
  int i = (blockIdx.x * blockDim.x + threadIdx.x) * 4;
  if (i + 3 < nelem) {
    f32x4 f = *(const f32x4*)(src + i);
    u32x2 p;
    p.x = pack2_bf16(f.x, f.y);
    p.y = pack2_bf16(f.z, f.w);
    *(u32x2*)(dst + i) = p;
  }
}

// src[K,N] f32 row-major  ->  dst[N,K] bf16 row-major (weight transpose)
__global__ void __launch_bounds__(256)
k_cvt_transpose(const float* __restrict__ src, unsigned short* __restrict__ dst,
                int K, int N) {
  int t = blockIdx.x * blockDim.x + threadIdx.x;
  if (t >= K * N) return;
  int j = t / K;
  int k = t - j * K;
  dst[(size_t)j * K + k] = f32_to_bf16_bits(src[(size_t)k * N + j]);
}

// -------------------- q projection: [2048,768] x [768,768]^T --------------------
__global__ void __launch_bounds__(128)
k_gemm_qproj(const unsigned short* __restrict__ A, const unsigned short* __restrict__ Bt,
             unsigned short* __restrict__ qout /* [BH, n, D] */) {
  const int lane = threadIdx.x & 31, wid = threadIdx.x >> 5;
  const int m0 = blockIdx.x * 16;
  const int n0 = blockIdx.y * 256 + wid * 64;
  ACC_INIT;
  gemm_wave<false>(A, TC, Bt, TC, m0, n0, TC, lane, acc);
#pragma unroll
  for (int t = 0; t < 4; ++t) {
    int cc = n0 + t * 16 + (lane & 15);
    int h = cc >> 6, d = cc & 63;
#pragma unroll
    for (int r = 0; r < 8; ++r) {
      int m = m0 + r + ((lane & 16) ? 8 : 0);
      int b = m >> 10, i = m & 1023;
      qout[(((size_t)(b * THH + h) * TNQ + i) << 6) + d] = f32_to_bf16_bits(acc[t][r]);
    }
  }
}

// -------------------- kv projection: [8192,768] x [768,1536] --------------------
// K (reduction) = C = 768; output width = 2C = 1536 handled by the grid.
__global__ void __launch_bounds__(128)
k_gemm_kvproj(const unsigned short* __restrict__ A, const unsigned short* __restrict__ Bt,
              unsigned short* __restrict__ kout /* [BH, N, D] */,
              unsigned short* __restrict__ vTout /* [BH, D, N] */) {
  const int lane = threadIdx.x & 31, wid = threadIdx.x >> 5;
  const int m0 = blockIdx.x * 16;
  const int n0 = blockIdx.y * 256 + wid * 64;
  ACC_INIT;
  gemm_wave<false>(A, TC, Bt, TC, m0, n0, TC, lane, acc);
#pragma unroll
  for (int t = 0; t < 4; ++t) {
    int cc = n0 + t * 16 + (lane & 15);
#pragma unroll
    for (int r = 0; r < 8; ++r) {
      int m = m0 + r + ((lane & 16) ? 8 : 0);
      int b = m >> 12, j = m & 4095;
      unsigned short bits = f32_to_bf16_bits(acc[t][r]);
      if (cc < TC) {          // k
        int h = cc >> 6, d = cc & 63;
        kout[(((size_t)(b * THH + h) * TNK + j) << 6) + d] = bits;
      } else {                // v (store transposed)
        int c2 = cc - TC;
        int h = c2 >> 6, d = c2 & 63;
        vTout[((size_t)((b * THH + h) * TD + d) << 12) + j] = bits;
      }
    }
  }
}

// -------------------- scores: S = scale * q k^T -> f32 xattn region --------------------
__global__ void __launch_bounds__(128)
k_scores(const unsigned short* __restrict__ qb, const unsigned short* __restrict__ kb,
         float* __restrict__ S) {
  const int lane = threadIdx.x & 31, wid = threadIdx.x >> 5;
  const int bh = blockIdx.z;
  const unsigned short* q = qb + (size_t)bh * TNQ * TD;
  const unsigned short* k = kb + (size_t)bh * TNK * TD;
  float* s = S + (size_t)bh * TNQ * TNK;
  const int m0 = blockIdx.x * 16;
  const int n0 = blockIdx.y * 256 + wid * 64;
  ACC_INIT;
  gemm_wave<false>(q, TD, k, TD, m0, n0, TD, lane, acc);
  const float scale = 0.125f;   // D^-0.5, D=64
#pragma unroll
  for (int t = 0; t < 4; ++t) {
    int cc = n0 + t * 16 + (lane & 15);
#pragma unroll
    for (int r = 0; r < 8; ++r) {
      int m = m0 + r + ((lane & 16) ? 8 : 0);
      s[(size_t)m * TNK + cc] = acc[t][r] * scale;
    }
  }
}

// -------------------- row softmax over 4096, in place --------------------
// Stages the row into LDS with gfx1250 async global->LDS copies (ASYNCcnt
// path), then reduces from LDS.
__global__ void __launch_bounds__(256)
k_softmax(float* __restrict__ S) {
  __shared__ float lrow[TNK];
  __shared__ float red[256];
  const int tid = threadIdx.x;
  float* p = S + (size_t)blockIdx.x * TNK;

  {
    // Thread t stages columns [16t, 16t+16) via 4 async b128 copies.
    const float* g = p + tid * 16;
    // Low 32 bits of the generic shared-pointer == LDS byte offset
    // (FLAT aperture mapping).
    unsigned lbase = (unsigned)(uintptr_t)(&lrow[0]) + (unsigned)(tid * 64);
#pragma unroll
    for (int c = 0; c < 4; ++c) {
      asm volatile("global_load_async_to_lds_b128 %0, %1, off"
                   :: "v"(lbase + c * 16), "v"(g + c * 4)
                   : "memory");
    }
    asm volatile("s_wait_asynccnt 0" ::: "memory");
  }
  __syncthreads();   // every wave has waited on its own ASYNCcnt

  float vals[16];
  float mx = -3.402823466e38f;
#pragma unroll
  for (int i = 0; i < 16; ++i) { vals[i] = lrow[tid + (i << 8)]; mx = fmaxf(mx, vals[i]); }
  red[tid] = mx; __syncthreads();
  for (int s = 128; s > 0; s >>= 1) {
    if (tid < s) red[tid] = fmaxf(red[tid], red[tid + s]);
    __syncthreads();
  }
  mx = red[0]; __syncthreads();
  float sum = 0.f;
#pragma unroll
  for (int i = 0; i < 16; ++i) { vals[i] = __expf(vals[i] - mx); sum += vals[i]; }
  red[tid] = sum; __syncthreads();
  for (int s = 128; s > 0; s >>= 1) {
    if (tid < s) red[tid] += red[tid + s];
    __syncthreads();
  }
  float inv = 1.0f / red[0];
#pragma unroll
  for (int i = 0; i < 16; ++i) p[tid + (i << 8)] = vals[i] * inv;
}

// -------------------- PV: O = P(f32->bf16) @ V -> o_bf [B*n, C] --------------------
__global__ void __launch_bounds__(128)
k_pv(const float* __restrict__ P, const unsigned short* __restrict__ vT,
     unsigned short* __restrict__ obf) {
  const int lane = threadIdx.x & 31, wid = threadIdx.x >> 5;
  const int bh = blockIdx.y;
  const float* p = P + (size_t)bh * TNQ * TNK;
  const unsigned short* v = vT + (size_t)bh * TD * TNK;
  const int m0 = blockIdx.x * 64 + wid * 16;
  ACC_INIT;
  gemm_wave<true>(p, TNK, v, TNK, m0, 0, TNK, lane, acc);
  const int b = bh / THH, h = bh % THH;
#pragma unroll
  for (int t = 0; t < 4; ++t) {
    int d = t * 16 + (lane & 15);
#pragma unroll
    for (int r = 0; r < 8; ++r) {
      int i = m0 + r + ((lane & 16) ? 8 : 0);
      obf[((size_t)(b * TNQ + i)) * TC + h * TD + d] = f32_to_bf16_bits(acc[t][r]);
    }
  }
}

// -------------------- out projection + bias: f32 output --------------------
__global__ void __launch_bounds__(128)
k_outproj(const unsigned short* __restrict__ obf, const unsigned short* __restrict__ WpT,
          const float* __restrict__ bias, float* __restrict__ out) {
  const int lane = threadIdx.x & 31, wid = threadIdx.x >> 5;
  const int m0 = blockIdx.x * 16;
  const int n0 = blockIdx.y * 256 + wid * 64;
  ACC_INIT;
  gemm_wave<false>(obf, TC, WpT, TC, m0, n0, TC, lane, acc);
#pragma unroll
  for (int t = 0; t < 4; ++t) {
    int cc = n0 + t * 16 + (lane & 15);
    float bb = bias[cc];
#pragma unroll
    for (int r = 0; r < 8; ++r) {
      int m = m0 + r + ((lane & 16) ? 8 : 0);
      out[(size_t)m * TC + cc] = acc[t][r] + bb;
    }
  }
}

// -------------------- host launcher --------------------
extern "C" void kernel_launch(void* const* d_in, const int* in_sizes, int n_in,
                              void* d_out, int out_size, void* d_ws, size_t ws_size,
                              hipStream_t stream) {
  const float* q_in  = (const float*)d_in[0];
  const float* x     = (const float*)d_in[1];
  const float* Wq    = (const float*)d_in[2];
  const float* Wkv   = (const float*)d_in[3];
  const float* Wproj = (const float*)d_in[4];
  const float* bproj = (const float*)d_in[5];

  float* out   = (float*)d_out;                       // [B, n, C]
  float* xattn = out + (size_t)TB * TNQ * TC;         // [B, H, n, N]

  // workspace carve-up (256B aligned regions)
  char* w = (char*)d_ws;
  auto carve = [&](size_t bytes) {
    char* p = w;
    w += (bytes + 255) & ~(size_t)255;
    return p;
  };
  unsigned short* qin_bf = (unsigned short*)carve((size_t)TB * TNQ * TC * 2);
  unsigned short* x_bf   = (unsigned short*)carve((size_t)TB * TNK * TC * 2);
  unsigned short* WqT    = (unsigned short*)carve((size_t)TC * TC * 2);
  unsigned short* WkvT   = (unsigned short*)carve((size_t)2 * TC * TC * 2);
  unsigned short* WpT    = (unsigned short*)carve((size_t)TC * TC * 2);
  unsigned short* q_bf   = (unsigned short*)carve((size_t)TBH * TNQ * TD * 2);
  unsigned short* k_bf   = (unsigned short*)carve((size_t)TBH * TNK * TD * 2);
  unsigned short* vT_bf  = (unsigned short*)carve((size_t)TBH * TD * TNK * 2);
  unsigned short* o_bf   = (unsigned short*)carve((size_t)TB * TNQ * TC * 2);

  // 1) convert activations to bf16; transpose weights to [N,K] bf16
  {
    int ne_q = TB * TNQ * TC;       // 1,572,864
    int ne_x = TB * TNK * TC;       // 6,291,456
    k_cvt_bf16<<<dim3(ne_q / 1024), 256, 0, stream>>>(q_in, qin_bf, ne_q);
    k_cvt_bf16<<<dim3(ne_x / 1024), 256, 0, stream>>>(x, x_bf, ne_x);
    k_cvt_transpose<<<dim3((TC * TC) / 256), 256, 0, stream>>>(Wq, WqT, TC, TC);
    k_cvt_transpose<<<dim3((TC * 2 * TC) / 256), 256, 0, stream>>>(Wkv, WkvT, TC, 2 * TC);
    k_cvt_transpose<<<dim3((TC * TC) / 256), 256, 0, stream>>>(Wproj, WpT, TC, TC);
  }

  // 2) projections (bf16 WMMA)
  k_gemm_qproj<<<dim3((TB * TNQ) / 16, TC / 256), 128, 0, stream>>>(qin_bf, WqT, q_bf);
  k_gemm_kvproj<<<dim3((TB * TNK) / 16, (2 * TC) / 256), 128, 0, stream>>>(x_bf, WkvT, k_bf, vT_bf);

  // 3) attention scores into xattn region of d_out
  k_scores<<<dim3(TNQ / 16, TNK / 256, TBH), 128, 0, stream>>>(q_bf, k_bf, xattn);

  // 4) row softmax in place (B*H*n rows of 4096), async-staged through LDS
  k_softmax<<<dim3(TBH * TNQ), 256, 0, stream>>>(xattn);

  // 5) O = P @ V (A from f32 probabilities, converted in-register)
  k_pv<<<dim3(TNQ / 64, TBH), 128, 0, stream>>>(xattn, vT_bf, o_bf);

  // 6) output projection + bias -> f32 out region
  k_outproj<<<dim3((TB * TNQ) / 16, TC / 256), 128, 0, stream>>>(o_bf, WpT, bproj, out);
}
